// ATDTransformerLayerGeo_35098472743143
// MI455X (gfx1250) — matrix operand order repaired and checked by reading, not compiled, not run on hardware
//
#include <hip/hip_runtime.h>
#include <cmath>

typedef __attribute__((ext_vector_type(16))) _Float16 v16h;
typedef __attribute__((ext_vector_type(8)))  float    v8f;
typedef __attribute__((ext_vector_type(4)))  _Float16 h4;

union V8    { v8f  v; float f[8]; };
union FragU { v16h h; uint4 u[2]; };

#if defined(__has_builtin)
#if __has_builtin(__builtin_amdgcn_s_wait_asynccnt)
#define HAVE_WAIT_ASYNC 1
#endif
#endif

// ---- WMMA fragment loaders (CDNA5 ISA §7.12.2) ----------------
// A-matrix 16x32 f16: lane l holds row m=l&15; K halves [kb..kb+7],[kb+16..kb+23],
// kb = 0 (lanes 0-15) or 8 (lanes 16-31). Source is row-major [M][K] (halfs).
__device__ __forceinline__ v16h load_afrag(const _Float16* base, int mBase,
                                           int strideHalfs, int kOff) {
  const int lane = threadIdx.x & 31;
  const int m    = lane & 15;
  const int kb   = kOff + ((lane & 16) ? 8 : 0);
  const _Float16* p = base + (size_t)(mBase + m) * strideHalfs + kb;
  FragU r;
  r.u[0] = *(const uint4*)(p);
  r.u[1] = *(const uint4*)(p + 16);
  return r.h;
}
// B-matrix 32x16 f16: lane l holds col n=l&15; K halves [kb..kb+7],[kb+16..kb+23],
// kb = 0 (lanes 0-15) or 8 (lanes 16-31). Source is row-major [N][K] (halfs),
// i.e. B stored transposed (column n contiguous over K).
__device__ __forceinline__ v16h load_bfrag(const _Float16* base, int nBase,
                                           int strideHalfs, int kOff) {
  const int lane = threadIdx.x & 31;
  const int n    = lane & 15;
  const int kb   = kOff + ((lane & 16) ? 8 : 0);
  const _Float16* p = base + (size_t)(nBase + n) * strideHalfs + kb;
  FragU r;
  r.u[0] = *(const uint4*)(p);
  r.u[1] = *(const uint4*)(p + 16);
  return r.h;
}

// ---- constants ------------------------------------------------------------
// B=2 H=W=192 C=192 WS=16 SHIFT=8 NH=6 HD=32 N=256; 12x12=144 windows/image.

// ---- kernel 0: per-window cumulative geometric terms ct_h / ct_v ----------
__global__ __launch_bounds__(256) void ct_kernel(const float* __restrict__ x,
                                                 const float* __restrict__ geo_sigma,
                                                 float2* __restrict__ cthvG) {
  __shared__ float dhs[256], dvs[256], cth[256], ctv[256];
  const int tid = threadIdx.x;
  const int win = blockIdx.x;
  const int b = win / 144, wy = (win % 144) / 12, wx = win % 12;
  const int r = tid >> 4, c = tid & 15;
  const int ys  = (wy * 16 + r + 8) % 192;
  const int xs  = (wx * 16 + c + 8) % 192;
  const int ysu = (wy * 16 + r - 1 + 8) % 192;   // only used when r>0 (nonneg)
  const int xsl = (wx * 16 + c - 1 + 8) % 192;   // only used when c>0 (nonneg)
  const float4* self4 = (const float4*)(x + (size_t)((b * 192 + ys) * 192 + xs) * 192);
  const float4* left4 = (const float4*)(x + (size_t)((b * 192 + ys) * 192 + xsl) * 192);
  const float4* up4   = (const float4*)(x + (size_t)((b * 192 + ysu) * 192 + xs) * 192);
  float sh = 0.f, sv = 0.f;
  for (int d4 = 0; d4 < 48; ++d4) {
    const float4 s = self4[d4];
    const float4 l = left4[d4];
    const float4 u = up4[d4];
    sh += fabsf(s.x - l.x) + fabsf(s.y - l.y) + fabsf(s.z - l.z) + fabsf(s.w - l.w);
    sv += fabsf(s.x - u.x) + fabsf(s.y - u.y) + fabsf(s.z - u.z) + fabsf(s.w - u.w);
  }
  const float sigma = fabsf(geo_sigma[0]);
  dhs[tid] = (c == 0) ? 0.f : 1.f + sigma * sh;
  dvs[tid] = (r == 0) ? 0.f : 1.f + sigma * sv;
  __syncthreads();
  if (tid < 16) {              // inclusive row scan -> ct_h
    float a = 0.f;
    for (int cc = 0; cc < 16; ++cc) { a += dhs[tid * 16 + cc]; cth[tid * 16 + cc] = a; }
  } else if (tid < 32) {       // inclusive column scan -> ct_v
    const int cc = tid - 16;
    float a = 0.f;
    for (int rr = 0; rr < 16; ++rr) { a += dvs[rr * 16 + cc]; ctv[rr * 16 + cc] = a; }
  }
  __syncthreads();
  cthvG[(size_t)win * 256 + tid] = make_float2(cth[tid], ctv[tid]);
}

// ---- kernel 0b: gather rpb_full[h][i][j] = rpb_table[rpi[i,j], h] ---------
__global__ __launch_bounds__(256) void rpb_kernel(const float* __restrict__ rpb_table,
                                                  const int* __restrict__ rpi,
                                                  float* __restrict__ rpbF) {
  const int e = blockIdx.x * 256 + threadIdx.x;   // 0..65535
  const int idx = rpi[e];
  #pragma unroll
  for (int h = 0; h < 6; ++h)
    rpbF[(size_t)h * 65536 + e] = rpb_table[idx * 6 + h];
}

// ---- kernel 0c: f16 copy of proj_w (row n = output channel, K contiguous) -
__global__ __launch_bounds__(256) void wcvt_kernel(const float* __restrict__ proj_w,
                                                   unsigned short* __restrict__ wf16raw) {
  const int e = blockIdx.x * 256 + threadIdx.x;   // 0..36863 (144 blocks)
  const _Float16 h = (_Float16)proj_w[e];
  wf16raw[e] = *(const unsigned short*)&h;
}

// ---- kernel 1: windowed attention (per window, per head, per 128-row half)
__global__ __launch_bounds__(256) void attn_kernel(const float* __restrict__ qkv,
                                                   const float2* __restrict__ cthvG,
                                                   const float* __restrict__ rpbF,
                                                   const float* __restrict__ attn_mask,
                                                   const float* __restrict__ geo_scale,
                                                   unsigned short* __restrict__ Oraw) {
  __shared__ __align__(16) _Float16 Klds[256 * 32];     // 16 KB  K row-major
  __shared__ __align__(16) _Float16 Vt[32 * 256];       // 16 KB  V transposed
  __shared__ __align__(16) _Float16 Qlds[128 * 32];     //  8 KB  Q (scaled)
  __shared__ __align__(16) _Float16 Plds[8 * 16 * 256]; // 64 KB  per-wave P
  __shared__ float2 cthv_s[256];                        //  2 KB

  const int tid = threadIdx.x;
  const int rowhalf = blockIdx.x;
  const int head = blockIdx.y;
  const int win = blockIdx.z;
  const int b = win / 144, wy = (win % 144) / 12, wx = win % 12;
  _Float16* Oscr = (_Float16*)Oraw;

  // ---- stage K, V (full window, this head) -------------------------------
  {
    const int r = tid >> 4, c = tid & 15;
    const int ys = (wy * 16 + r + 8) % 192;
    const int xs = (wx * 16 + c + 8) % 192;
    const float4* q4 = (const float4*)(qkv + (size_t)((b * 192 + ys) * 192 + xs) * 576);
    #pragma unroll
    for (int d4 = 0; d4 < 8; ++d4) {
      const float4 kf = q4[48 + head * 8 + d4];  // +192 floats = K block
      h4 kh = { (_Float16)kf.x, (_Float16)kf.y, (_Float16)kf.z, (_Float16)kf.w };
      *(h4*)(&Klds[tid * 32 + d4 * 4]) = kh;
      const float4 vf = q4[96 + head * 8 + d4];  // +384 floats = V block
      Vt[(d4 * 4 + 0) * 256 + tid] = (_Float16)vf.x;
      Vt[(d4 * 4 + 1) * 256 + tid] = (_Float16)vf.y;
      Vt[(d4 * 4 + 2) * 256 + tid] = (_Float16)vf.z;
      Vt[(d4 * 4 + 3) * 256 + tid] = (_Float16)vf.w;
    }
  }
  // ---- stage Q rows of this half, pre-scaled by hd^-0.5 ------------------
  if (tid < 128) {
    const int tok = rowhalf * 128 + tid;
    const int r = tok >> 4, c = tok & 15;
    const int ys = (wy * 16 + r + 8) % 192;
    const int xs = (wx * 16 + c + 8) % 192;
    const float4* q4 = (const float4*)(qkv + (size_t)((b * 192 + ys) * 192 + xs) * 576);
    const float scale = 0.17677669529663687f;  // 32^-0.5
    #pragma unroll
    for (int d4 = 0; d4 < 8; ++d4) {
      const float4 qf = q4[head * 8 + d4];
      h4 qh = { (_Float16)(qf.x * scale), (_Float16)(qf.y * scale),
                (_Float16)(qf.z * scale), (_Float16)(qf.w * scale) };
      *(h4*)(&Qlds[tid * 32 + d4 * 4]) = qh;
    }
  }
  cthv_s[tid] = cthvG[(size_t)win * 256 + tid];
  __syncthreads();

  const int w = tid >> 5;
  const int lane = tid & 31;
  const int lane15 = lane & 15;
  const int hadd = (lane & 16) ? 8 : 0;    // C/D layout: M = v + hadd
  const int i0 = rowhalf * 128 + w * 16;   // first window-row of this wave

  // ---- S = (Q*scale) K^T : 16 WMMAs (K=32 = one head-dim) ----------------
  V8 S[16];
  const v16h aQ = load_afrag(Qlds, w * 16, 32, 0);
  #pragma unroll
  for (int t = 0; t < 16; ++t) {
    const v16h bK = load_bfrag(Klds, t * 16, 32, 0);
    v8f cz = {};
    S[t].v = __builtin_amdgcn_wmma_f32_16x16x32_f16(false, aQ, false, bK,
                                                    (short)0, cz, false, false);
  }

  // ---- biases + running row max in C/D register layout -------------------
  const float* rpbH = rpbF + (size_t)head * 65536;
  const float* mskW = attn_mask + (size_t)(win % 144) * 65536;
  const float gs = geo_scale[head];
  float chf[8], crh[8], crv[8];
  #pragma unroll
  for (int v = 0; v < 8; ++v) {
    const int i = i0 + v + hadd;                 // window-row index 0..255
    const float2 cs = cthv_s[i];
    chf[v] = cs.x;                               // ct_h flat(i)
    const float2 cc = cthv_s[(i & 0xF0) | lane15];  // corner [r_i, c_j]
    crh[v] = cc.x; crv[v] = cc.y;
  }
  float rmax[8];
  #pragma unroll
  for (int v = 0; v < 8; ++v) rmax[v] = -3.0e38f;

  auto biasPass = [&](bool useMask) {
    #pragma unroll
    for (int t = 0; t < 16; ++t) {
      const int j = t * 16 + lane15;
      const float cvf = cthv_s[j].y;             // ct_v flat(j)
      #pragma unroll
      for (int v = 0; v < 8; ++v) {
        const int i = i0 + v + hadd;
        const float geo = fabsf(crh[v] - chf[v]) + fabsf(cvf - crv[v]);
        float val = S[t].f[v] + rpbH[i * 256 + j] - geo * gs;
        if (useMask) val += mskW[i * 256 + j];
        S[t].f[v] = val;
        rmax[v] = fmaxf(rmax[v], val);
      }
    }
  };
  // only windows touching the rolled boundary have a non-zero shift mask
  if (wy == 11 || wx == 11) biasPass(true); else biasPass(false);

  #pragma unroll
  for (int m = 1; m <= 8; m <<= 1)
    #pragma unroll
    for (int v = 0; v < 8; ++v)
      rmax[v] = fmaxf(rmax[v], __shfl_xor(rmax[v], m, 32));
  float rsum[8];
  #pragma unroll
  for (int v = 0; v < 8; ++v) rsum[v] = 0.f;
  #pragma unroll
  for (int t = 0; t < 16; ++t)
    #pragma unroll
    for (int v = 0; v < 8; ++v) {
      const float p = exp2f((S[t].f[v] - rmax[v]) * 1.4426950408889634f);
      S[t].f[v] = p;
      rsum[v] += p;
    }
  #pragma unroll
  for (int m = 1; m <= 8; m <<= 1)
    #pragma unroll
    for (int v = 0; v < 8; ++v)
      rsum[v] += __shfl_xor(rsum[v], m, 32);
  float rinv[8];
  #pragma unroll
  for (int v = 0; v < 8; ++v) rinv[v] = 1.0f / rsum[v];

  // ---- P -> LDS (f16, row-major) to re-fragment as A-matrix --------------
  _Float16* Pw = Plds + w * 16 * 256;
  #pragma unroll
  for (int t = 0; t < 16; ++t)
    #pragma unroll
    for (int v = 0; v < 8; ++v)
      Pw[(v + hadd) * 256 + t * 16 + lane15] = (_Float16)(S[t].f[v] * rinv[v]);

  // ---- O = P V : 2 N-tiles x 8 K-chunks of 32; store f16 scratch ---------
  #pragma unroll
  for (int nt = 0; nt < 2; ++nt) {
    V8 o; v8f z = {}; o.v = z;
    #pragma unroll
    for (int kt = 0; kt < 8; ++kt) {
      const v16h aP = load_afrag(Pw, 0, 256, kt * 32);
      const v16h bV = load_bfrag(Vt, nt * 16, 256, kt * 32);
      o.v = __builtin_amdgcn_wmma_f32_16x16x32_f16(false, aP, false, bV,
                                                   (short)0, o.v, false, false);
    }
    #pragma unroll
    for (int v = 0; v < 8; ++v) {
      const int i = i0 + v + hadd;
      Oscr[((size_t)win * 256 + i) * 192 + head * 32 + nt * 16 + lane15] =
          (_Float16)o.f[v];
    }
  }
}

// ---- kernel 2: output projection + window-reverse + roll ------------------
__global__ __launch_bounds__(512) void proj_kernel(const unsigned short* __restrict__ Oraw,
                                                   const unsigned short* __restrict__ wf16raw,
                                                   const float* __restrict__ proj_b,
                                                   float* __restrict__ out) {
  __shared__ __align__(16) _Float16 Olds[256 * 192];   // 96 KB
  const int tid = threadIdx.x;
  const int win = blockIdx.x;
  const _Float16* wf16 = (const _Float16*)wf16raw;
  const _Float16* Osrc = (const _Float16*)Oraw + (size_t)win * 49152;

  // ---- stage O tile: async global->LDS b128 copies (ASYNCcnt-tracked) ----
  // 512 lanes x 16 B x 12 iterations = 96 KB. LDS operand = low 32 bits of
  // the flat pointer (flat->LDS truncation, ISA 10.2); per-iteration stride
  // folded into the signed-24-bit instruction offset.
  {
    const unsigned long long ga =
        (unsigned long long)Osrc + (unsigned long long)tid * 16ull;
    const unsigned la =
        (unsigned)(unsigned long long)&Olds[0] + (unsigned)tid * 16u;
    #pragma unroll
    for (int it = 0; it < 12; ++it) {
      asm volatile("global_load_async_to_lds_b128 %0, %1, off offset:%2"
                   :: "v"(la), "v"(ga), "i"(it * 8192)
                   : "memory");
    }
  }
#if defined(HAVE_WAIT_ASYNC)
  __builtin_amdgcn_s_wait_asynccnt(0);
#else
  asm volatile("s_wait_asynccnt 0x0" ::: "memory");
#endif
  __syncthreads();

  const int w = tid >> 5;
  const int lane = tid & 31;
  const int lane15 = lane & 15;
  const int hadd = (lane & 16) ? 8 : 0;
  const int m0 = w * 16;                       // 16 waves x 16 tokens = 256
  const int b = win / 144, wy = (win % 144) / 12, wx = win % 12;

  v16h afr[6];
  #pragma unroll
  for (int kc = 0; kc < 6; ++kc) afr[kc] = load_afrag(Olds, m0, 192, kc * 32);

  #pragma unroll
  for (int nt = 0; nt < 12; ++nt) {
    V8 o; v8f z = {}; o.v = z;
    #pragma unroll
    for (int kc = 0; kc < 6; ++kc) {
      const v16h bw = load_bfrag(wf16, nt * 16, 192, kc * 32);  // global (L2)
      o.v = __builtin_amdgcn_wmma_f32_16x16x32_f16(false, afr[kc], false, bw,
                                                   (short)0, o.v, false, false);
    }
    const int ch = nt * 16 + lane15;
    const float pb = proj_b[ch];
    #pragma unroll
    for (int v = 0; v < 8; ++v) {
      const int tok = m0 + v + hadd;
      const int p = (wy * 16 + (tok >> 4) + 8) % 192;   // un-shift roll
      const int q = (wx * 16 + (tok & 15) + 8) % 192;
      out[((size_t)b * 36864 + p * 192 + q) * 192 + ch] = o.f[v] + pb;
    }
  }
}

// ---- host launcher --------------------------------------------------------
extern "C" void kernel_launch(void* const* d_in, const int* in_sizes, int n_in,
                              void* d_out, int out_size, void* d_ws, size_t ws_size,
                              hipStream_t stream) {
  const float* x          = (const float*)d_in[0];
  const float* qkv        = (const float*)d_in[1];
  const float* rpb_table  = (const float*)d_in[2];
  const float* geo_scale  = (const float*)d_in[3];
  const float* geo_sigma  = (const float*)d_in[4];
  const float* proj_w     = (const float*)d_in[5];
  const float* proj_b     = (const float*)d_in[6];
  const int*   rpi        = (const int*)d_in[7];
  const float* attn_mask  = (const float*)d_in[8];
  float* out = (float*)d_out;
  (void)in_sizes; (void)n_in; (void)out_size; (void)ws_size;

  char* ws = (char*)d_ws;
  size_t off = 0;
  auto alloc = [&](size_t bytes) {
    char* p = ws + off;
    off = (off + bytes + 255) & ~(size_t)255;
    return p;
  };
  float2*         cthvG = (float2*)alloc((size_t)288 * 256 * sizeof(float2));
  float*          rpbF  = (float*)alloc((size_t)6 * 65536 * sizeof(float));
  unsigned short* wf16  = (unsigned short*)alloc((size_t)36864 * sizeof(unsigned short));
  unsigned short* Oscr  = (unsigned short*)alloc((size_t)288 * 256 * 192 * sizeof(unsigned short));

  ct_kernel<<<288, 256, 0, stream>>>(x, geo_sigma, cthvG);
  rpb_kernel<<<256, 256, 0, stream>>>(rpb_table, rpi, rpbF);
  wcvt_kernel<<<144, 256, 0, stream>>>(proj_w, wf16);
  attn_kernel<<<dim3(2, 6, 288), 256, 0, stream>>>(qkv, cthvG, rpbF, attn_mask,
                                                   geo_scale, Oscr);
  proj_kernel<<<288, 512, 0, stream>>>(Oscr, wf16, proj_b, out);
}
